// edge_gnn_80212809220217
// MI455X (gfx1250) — compile-verified, matrix operand order checked by reading, not run on hardware
//
#include <hip/hip_runtime.h>

// ---------------------------------------------------------------------------
// Fused edge-GNN for gfx1250 (MI455X): f16 WMMA (f32 accum), padded f16 LDS
// staging (bank-conflict-free), f16 node-embedding buffer, f32 LN statistics,
// sched_barrier-pinned double-buffered B pipeline, preloaded fragment biases.
// N=50000 nodes, E=400000 edges, HC=HD=128.
// ---------------------------------------------------------------------------

typedef __attribute__((ext_vector_type(16))) _Float16 v16h;
typedef __attribute__((ext_vector_type(8)))  _Float16 v8h;
typedef __attribute__((ext_vector_type(8)))  float    v8f;
typedef __attribute__((ext_vector_type(4)))  float    v4f;

#define HD  128
#define SST 136           // LDS stage row stride in halves (272B -> +4 banks/row)
#define NW  5             // waves per block
#define BLK (NW * 32)     // threads per block

#if defined(__has_builtin)
#  if __has_builtin(__builtin_amdgcn_sched_barrier)
#    define SCHED_FENCE() __builtin_amdgcn_sched_barrier(0)
#  endif
#endif
#ifndef SCHED_FENCE
#  define SCHED_FENCE()
#endif

static __device__ __forceinline__ v8f wmma32(v16h a, v16h b, v8f c) {
    return __builtin_amdgcn_wmma_f32_16x16x32_f16(false, a, false, b, (short)0, c,
                                                  false, false);
}

// Load one f16 fragment (per-lane CDNA5 16-bit 16x32 layout) from a row base:
// elements 0..7 = cols kb..kb+7, elements 8..15 = cols kb+16..kb+23.
static __device__ __forceinline__ v16h load_frag16(const _Float16* base, int kb) {
    v8h lo = *(const v8h*)(base + kb);
    v8h hi = *(const v8h*)(base + kb + 16);
    v16h r;
#pragma unroll
    for (int i = 0; i < 8; ++i) { r[i] = lo[i]; r[8 + i] = hi[i]; }
    return r;
}

// Gather a 16x128 A-fragment set from the f16 node-embedding buffer (no cvt).
static __device__ __forceinline__ void gather_afrags_h(const _Float16* H, int rid,
                                                       int lane, v16h a[4]) {
    const int hb = (lane >> 4) << 3;
    const _Float16* base = H + (size_t)rid * HD;
#pragma unroll
    for (int t = 0; t < 4; ++t) a[t] = load_frag16(base, t * 32 + hb);
}

// Gather from a global f32 matrix with (x - mean) * inv_std normalization.
static __device__ __forceinline__ void gather_afrags_norm(const float* X, int rid,
                                                          const float* mean,
                                                          const float* inv,
                                                          int lane, v16h a[4]) {
    const int hb = (lane >> 4) << 3;
    const float* base = X + (size_t)rid * HD;
#pragma unroll
    for (int t = 0; t < 4; ++t) {
        const int kb = t * 32 + hb;
        v16h f;
#pragma unroll
        for (int c = 0; c < 2; ++c) {   // two 8-wide chunks: kb, kb+16
            const int o = kb + c * 16;
            v4f x0 = *(const v4f*)(base + o);
            v4f x1 = *(const v4f*)(base + o + 4);
            v4f m0 = *(const v4f*)(mean + o);
            v4f m1 = *(const v4f*)(mean + o + 4);
            v4f s0 = *(const v4f*)(inv + o);
            v4f s1 = *(const v4f*)(inv + o + 4);
#pragma unroll
            for (int i = 0; i < 4; ++i) {
                f[c * 8 + i]     = (_Float16)((x0[i] - m0[i]) * s0[i]);
                f[c * 8 + 4 + i] = (_Float16)((x1[i] - m1[i]) * s1[i]);
            }
        }
        a[t] = f;
    }
}

// Rebuild A fragments from the wave's padded f16 LDS stage (ds_load_b128,
// bank-conflict-free thanks to the 136-half row stride).
static __device__ __forceinline__ void load_afrags_lds(const _Float16* S, int lane,
                                                       v16h a[4]) {
    const int row = lane & 15;
    const int hb  = (lane >> 4) << 3;
    const _Float16* base = S + row * SST;
#pragma unroll
    for (int t = 0; t < 4; ++t) a[t] = load_frag16(base, t * 32 + hb);
}

// B fragment from f16 W^T (stored N x K row-major -> B column-major, identical
// per-lane pattern to the A fragment).
static __device__ __forceinline__ v16h load_bfrag(const _Float16* Wt, int K,
                                                  int ntile, int ktile, int lane) {
    const int nrow = ntile * 16 + (lane & 15);
    const int kb   = ktile * 32 + ((lane >> 4) << 3);
    return load_frag16(Wt + (size_t)nrow * K, kb);
}

// Load 4 B fragments (one full K-group for one N-tile) into a buffer.
static __device__ __forceinline__ void loadB4(v16h (&dst)[4], const _Float16* Wt,
                                              int K, int ntile, int kt0, int lane) {
#pragma unroll
    for (int k = 0; k < 4; ++k) dst[k] = load_bfrag(Wt, K, ntile, kt0 + k, lane);
}

static __device__ __forceinline__ v8f mm4(const v16h a[4], const v16h (&b)[4],
                                          v8f acc) {
#pragma unroll
    for (int k = 0; k < 4; ++k) acc = wmma32(a[k], b[k], acc);
    return acc;
}

// Per-lane fragment-ordered bias: 8 per-tile values in two contiguous b128
// loads, fetched ONCE per layer (keeps the inner loop free of loadcnt-0 waits).
static __device__ __forceinline__ v8f load_biasfrag(const float* biasR, int col0) {
    v4f lo = *(const v4f*)(biasR + col0 * 8);
    v4f hi = *(const v4f*)(biasR + col0 * 8 + 4);
    v8f b;
#pragma unroll
    for (int i = 0; i < 4; ++i) { b[i] = lo[i]; b[4 + i] = hi[i]; }
    return b;
}

static __device__ __forceinline__ v8f acc_init(v8f bvec, int n) {
    const float bv = bvec[n];
    v8f acc;
#pragma unroll
    for (int r = 0; r < 8; ++r) acc[r] = bv;
    return acc;
}

static __device__ __forceinline__ void store_tile(v8f acc, _Float16* S, int n,
                                                  int hb, int col0, bool relu) {
#pragma unroll
    for (int r = 0; r < 8; ++r) {
        float v = acc[r];
        if (relu) v = v > 0.f ? v : 0.f;
        S[((hb ? 8 : 0) + r) * SST + n * 16 + col0] = (_Float16)v;
    }
}

// One K=128 -> N=128 layer, unrolled by 2 with two B buffers. sched_barrier
// pins loads for tile n+1 ABOVE tile n's WMMAs, forcing overlapping live
// ranges (distinct registers) and a full WMMA-group of load-latency hiding.
static __device__ __forceinline__ void layer_k128(const v16h a[4],
                                                  const _Float16* Wt,
                                                  const float* biasR, _Float16* S,
                                                  int lane, bool relu) {
    const int hb   = lane >> 4;
    const int col0 = lane & 15;
    const v8f bvec = load_biasfrag(biasR, col0);
    v16h bA[4], bB[4];
    loadB4(bA, Wt, HD, 0, 0, lane);
#pragma unroll
    for (int nn = 0; nn < 4; ++nn) {
        const int n0 = 2 * nn, n1 = n0 + 1;
        loadB4(bB, Wt, HD, n1, 0, lane);
        SCHED_FENCE();
        v8f acc = acc_init(bvec, n0);
        acc = mm4(a, bA, acc);
        store_tile(acc, S, n0, hb, col0, relu);
        if (nn < 3) loadB4(bA, Wt, HD, n0 + 2, 0, lane);
        SCHED_FENCE();
        acc = acc_init(bvec, n1);
        acc = mm4(a, bB, acc);
        store_tile(acc, S, n1, hb, col0, relu);
    }
}

// K=384 layer over the implicit concat (own, nei, ea): linearized 24-step
// pipeline (8 N-tiles x 3 concat chunks), parity-alternating B buffers,
// schedule pinned with sched_barrier.
static __device__ __forceinline__ void layer_k384(const v16h a0[4],
                                                  const v16h a1[4],
                                                  const v16h a2[4],
                                                  const _Float16* Wt,
                                                  const float* biasR, _Float16* S,
                                                  int lane, bool relu) {
    const int hb   = lane >> 4;
    const int col0 = lane & 15;
    const v8f bvec = load_biasfrag(biasR, col0);
    v16h bA[4], bB[4];
    loadB4(bA, Wt, 384, 0, 0, lane);
    v8f acc;
#pragma unroll
    for (int s = 0; s < 24; ++s) {
        const int n = s / 3, c = s % 3;
        if (c == 0) acc = acc_init(bvec, n);
        const int ns = s + 1;
        const v16h* A = (c == 0) ? a0 : (c == 1) ? a1 : a2;
        if (s & 1) {
            if (s < 23) loadB4(bA, Wt, 384, ns / 3, (ns % 3) * 4, lane);
            SCHED_FENCE();
            acc = mm4(A, bB, acc);
        } else {
            if (s < 23) loadB4(bB, Wt, 384, ns / 3, (ns % 3) * 4, lane);
            SCHED_FENCE();
            acc = mm4(A, bA, acc);
        }
        if (c == 2) store_tile(acc, S, n, hb, col0, relu);
    }
}

// Final layer: accumulators written straight to global f32 (no stage, full
// f32 output precision). Same pinned double-buffered pipeline.
static __device__ __forceinline__ void layer_k128_out(const v16h a[4],
                                                      const _Float16* Wt,
                                                      const float* biasR,
                                                      float* out, int rowbase,
                                                      int limit, int lane) {
    const int hb   = lane >> 4;
    const int col0 = lane & 15;
    const v8f bvec = load_biasfrag(biasR, col0);
    v16h bA[4], bB[4];
    loadB4(bA, Wt, HD, 0, 0, lane);
#pragma unroll
    for (int nn = 0; nn < 4; ++nn) {
        const int n0 = 2 * nn, n1 = n0 + 1;
        loadB4(bB, Wt, HD, n1, 0, lane);
        SCHED_FENCE();
        v8f acc = acc_init(bvec, n0);
        acc = mm4(a, bA, acc);
#pragma unroll
        for (int r = 0; r < 8; ++r) {
            const int row = rowbase + (hb ? 8 : 0) + r;
            if (row < limit) out[(size_t)row * HD + n0 * 16 + col0] = acc[r];
        }
        if (nn < 3) loadB4(bA, Wt, HD, n0 + 2, 0, lane);
        SCHED_FENCE();
        acc = acc_init(bvec, n1);
        acc = mm4(a, bB, acc);
#pragma unroll
        for (int r = 0; r < 8; ++r) {
            const int row = rowbase + (hb ? 8 : 0) + r;
            if (row < limit) out[(size_t)row * HD + n1 * 16 + col0] = acc[r];
        }
    }
}

// Parallel LayerNorm of the f16 stage: 2 lanes per row, f32 statistics via a
// small LDS reduction. Caller must __syncthreads() after (cross-half reads).
static __device__ __forceinline__ void layernorm16(_Float16* S, float* red1,
                                                   float* red2,
                                                   const float* gamma,
                                                   const float* beta, int lane) {
    const int row  = lane & 15;
    const int half = lane >> 4;
    _Float16* p = S + row * SST + half * 64;
    float s1 = 0.f, s2 = 0.f;
#pragma unroll 16
    for (int c = 0; c < 64; ++c) {
        float v = (float)p[c];
        s1 += v; s2 += v * v;
    }
    red1[row * 2 + half] = s1;
    red2[row * 2 + half] = s2;
    __syncthreads();
    const float t1  = red1[row * 2] + red1[row * 2 + 1];
    const float t2  = red2[row * 2] + red2[row * 2 + 1];
    const float m   = t1 * (1.f / HD);
    const float var = t2 * (1.f / HD) - m * m;
    const float inv = rsqrtf(var + 1e-5f);
    const float* g = gamma + half * 64;
    const float* b = beta + half * 64;
#pragma unroll 16
    for (int c = 0; c < 64; ++c) {
        float v = (float)p[c];
        p[c] = (_Float16)((v - m) * inv * g[c] + b[c]);
    }
}

// ---------------------------------------------------------------------------
// Kernel 1: transpose + convert weights f32 (din x dout) -> f16 W^T (dout x din)
// ---------------------------------------------------------------------------
__global__ void gnn_prep_w(const float* __restrict__ W, _Float16* __restrict__ Wt,
                           int din, int dout) {
    int i = blockIdx.x * blockDim.x + threadIdx.x;
    int total = din * dout;
    if (i < total) {
        int k = i / dout, n = i % dout;
        Wt[(size_t)n * din + k] = (_Float16)W[(size_t)k * dout + n];
    }
}

// Kernel 1b: reciprocal of the std vectors (kills v_div chains in the hot path)
__global__ void gnn_prep_inv(const float* __restrict__ s, float* __restrict__ inv,
                             int n) {
    int i = blockIdx.x * blockDim.x + threadIdx.x;
    if (i < n) inv[i] = 1.f / s[i];
}

// Kernel 1c: reorganize a 128-float bias into per-lane fragment order:
// biasR[col0*8 + n] = bias[n*16 + col0]  (two b128 loads per lane per layer)
__global__ void gnn_prep_bias(const float* __restrict__ b,
                              float* __restrict__ bR) {
    int i = threadIdx.x;               // 0..127
    int col0 = i >> 3, n = i & 7;
    bR[i] = b[n * 16 + col0];
}

// ---------------------------------------------------------------------------
// Kernel 2: node encoder  h = LN(MLP3((x-mean)*inv_std))  -> f16 h buffer
// ---------------------------------------------------------------------------
struct NodeParams {
    const float* x;
    const float* mean;
    const float* inv;
    const _Float16 *w0, *w1, *w2;
    const float *b0, *b1, *b2;         // fragment-ordered biases
    const float *gamma, *beta;
    _Float16* h;
    int N;
};

__global__ __launch_bounds__(BLK) void gnn_node_kernel(NodeParams p) {
    __shared__ _Float16 S[NW][16 * SST];         // 5 * 4.25KB
    __shared__ float R1[NW][32], R2[NW][32];     // LN reductions
    const int lane = threadIdx.x & 31;
    const int wave = threadIdx.x >> 5;
    const int rowbase = (blockIdx.x * NW + wave) * 16;
    _Float16* Sw = &S[wave][0];

    int rid = rowbase + (lane & 15);
    if (rid >= p.N) rid = p.N - 1;

    v16h a[4];
    gather_afrags_norm(p.x, rid, p.mean, p.inv, lane, a);

    layer_k128(a, p.w0, p.b0, Sw, lane, true);
    __syncthreads(); load_afrags_lds(Sw, lane, a); __syncthreads();
    layer_k128(a, p.w1, p.b1, Sw, lane, true);
    __syncthreads(); load_afrags_lds(Sw, lane, a); __syncthreads();
    layer_k128(a, p.w2, p.b2, Sw, lane, false);
    __syncthreads();
    layernorm16(Sw, &R1[wave][0], &R2[wave][0], p.gamma, p.beta, lane);
    __syncthreads();

    // Coalesced f16 store of the 16x128 slab (as dwords).
    const unsigned* Sw32 = (const unsigned*)Sw;
#pragma unroll
    for (int r = 0; r < 16; ++r) {
        const int row = rowbase + r;
        if (row < p.N) {
            unsigned* H32 = (unsigned*)(p.h + (size_t)row * HD);
#pragma unroll
            for (int cc = 0; cc < 2; ++cc) {
                const int c = cc * 32 + lane;
                H32[c] = Sw32[r * (SST / 2) + c];
            }
        }
    }
}

// ---------------------------------------------------------------------------
// Kernel 3: fused per-edge pipeline (gather + edge_enc + edge_mlp + dec)
// ---------------------------------------------------------------------------
struct EdgeParams {
    const int* eidx;          // [2*E] int32
    const float* eattr;       // [E, 128]
    const float *mean_e, *inv_e;
    const _Float16* h;        // [N, 128] f16 encoded nodes
    const _Float16 *ee0, *ee1, *ee2;
    const _Float16 *em0, *em1, *em2, *em3;    // em0 is 128x384 (N x K)
    const _Float16 *dc0, *dc1, *dc2;
    const float *bee0, *bee1, *bee2;          // fragment-ordered biases
    const float *bem0, *bem1, *bem2, *bem3;
    const float *bdc0, *bdc1, *bdc2;
    const float *g_ee, *bt_ee, *g_em, *bt_em;
    float* out;               // [E, 128]
    int E;
};

__global__ __launch_bounds__(BLK) void gnn_edge_kernel(EdgeParams p) {
    __shared__ _Float16 S[NW][16 * SST];
    __shared__ float R1[NW][32], R2[NW][32];
    const int lane = threadIdx.x & 31;
    const int wave = threadIdx.x >> 5;
    const int ebase = (blockIdx.x * NW + wave) * 16;
    _Float16* Sw = &S[wave][0];

    // Warm L1/L2 for the first weight matrix (gfx1250 global_prefetch path).
    __builtin_prefetch((const char*)p.ee0 + (size_t)threadIdx.x * 128, 0, 1);

    int e = ebase + (lane & 15);
    if (e >= p.E) e = p.E - 1;
    const int own = p.eidx[e];
    const int nei = p.eidx[p.E + e];

    v16h fo[4], fn[4], fa[4];
    gather_afrags_h(p.h, own, lane, fo);          // x_own fragments (f16, no cvt)
    gather_afrags_h(p.h, nei, lane, fn);          // x_nei fragments
    gather_afrags_norm(p.eattr, e, p.mean_e, p.inv_e, lane, fa);

    // ---- edge encoder: LN(MLP3(ea)) ----
    layer_k128(fa, p.ee0, p.bee0, Sw, lane, true);
    __syncthreads(); load_afrags_lds(Sw, lane, fa); __syncthreads();
    layer_k128(fa, p.ee1, p.bee1, Sw, lane, true);
    __syncthreads(); load_afrags_lds(Sw, lane, fa); __syncthreads();
    layer_k128(fa, p.ee2, p.bee2, Sw, lane, false);
    __syncthreads();
    layernorm16(Sw, &R1[wave][0], &R2[wave][0], p.g_ee, p.bt_ee, lane);
    __syncthreads(); load_afrags_lds(Sw, lane, fa); __syncthreads();

    // ---- edge MLP: layer 1 over implicit concat(own, nei, ea), K = 384 ----
    layer_k384(fo, fn, fa, p.em0, p.bem0, Sw, lane, true);
    __syncthreads(); load_afrags_lds(Sw, lane, fa); __syncthreads();
    layer_k128(fa, p.em1, p.bem1, Sw, lane, true);
    __syncthreads(); load_afrags_lds(Sw, lane, fa); __syncthreads();
    layer_k128(fa, p.em2, p.bem2, Sw, lane, true);
    __syncthreads(); load_afrags_lds(Sw, lane, fa); __syncthreads();
    layer_k128(fa, p.em3, p.bem3, Sw, lane, false);
    __syncthreads();
    layernorm16(Sw, &R1[wave][0], &R2[wave][0], p.g_em, p.bt_em, lane);
    __syncthreads(); load_afrags_lds(Sw, lane, fa); __syncthreads();

    // ---- decoder ----
    layer_k128(fa, p.dc0, p.bdc0, Sw, lane, true);
    __syncthreads(); load_afrags_lds(Sw, lane, fa); __syncthreads();
    layer_k128(fa, p.dc1, p.bdc1, Sw, lane, true);
    __syncthreads(); load_afrags_lds(Sw, lane, fa); __syncthreads();
    layer_k128_out(fa, p.dc2, p.bdc2, p.out, ebase, p.E, lane);
}

// ---------------------------------------------------------------------------
// Host launcher
// ---------------------------------------------------------------------------
extern "C" void kernel_launch(void* const* d_in, const int* in_sizes, int n_in,
                              void* d_out, int out_size, void* d_ws, size_t ws_size,
                              hipStream_t stream) {
    // Input index map (pytree-flattened setup_inputs order)
    const float* x        = (const float*)d_in[0];
    const int*   eidx     = (const int*)d_in[1];      // jax default x64 off -> int32
    const float* eattr    = (const float*)d_in[2];
    const float* mean_x   = (const float*)d_in[3];
    const float* std_x    = (const float*)d_in[4];
    const float* mean_e   = (const float*)d_in[5];
    const float* std_e    = (const float*)d_in[6];
    const float* ne_W[3]  = {(const float*)d_in[7],  (const float*)d_in[9],  (const float*)d_in[11]};
    const float* ne_b[3]  = {(const float*)d_in[8],  (const float*)d_in[10], (const float*)d_in[12]};
    const float* ne_g     = (const float*)d_in[13];
    const float* ne_bt    = (const float*)d_in[14];
    const float* ee_W[3]  = {(const float*)d_in[15], (const float*)d_in[17], (const float*)d_in[19]};
    const float* ee_b[3]  = {(const float*)d_in[16], (const float*)d_in[18], (const float*)d_in[20]};
    const float* ee_g     = (const float*)d_in[21];
    const float* ee_bt    = (const float*)d_in[22];
    const float* em_W[4]  = {(const float*)d_in[23], (const float*)d_in[25], (const float*)d_in[27], (const float*)d_in[29]};
    const float* em_b[4]  = {(const float*)d_in[24], (const float*)d_in[26], (const float*)d_in[28], (const float*)d_in[30]};
    const float* em_g     = (const float*)d_in[31];
    const float* em_bt    = (const float*)d_in[32];
    const float* dc_W[3]  = {(const float*)d_in[33], (const float*)d_in[35], (const float*)d_in[37]};
    const float* dc_b[3]  = {(const float*)d_in[34], (const float*)d_in[36], (const float*)d_in[38]};

    const int N = in_sizes[0] / HD;          // 50000
    const int E = in_sizes[1] / 2;           // 400000

    // Workspace: 13 f16 W^T | 2 f32 inv-std | 13 fragment biases | f16 h buffer.
    _Float16* wts = (_Float16*)d_ws;
    size_t off = 0;
    _Float16* t_ne[3]; _Float16* t_ee[3]; _Float16* t_em[4]; _Float16* t_dc[3];
    for (int i = 0; i < 3; ++i) { t_ne[i] = wts + off; off += HD * HD; }
    for (int i = 0; i < 3; ++i) { t_ee[i] = wts + off; off += HD * HD; }
    t_em[0] = wts + off; off += 384 * HD;
    for (int i = 1; i < 4; ++i) { t_em[i] = wts + off; off += HD * HD; }
    for (int i = 0; i < 3; ++i) { t_dc[i] = wts + off; off += HD * HD; }
    size_t byte_off = off * sizeof(_Float16);
    float* inv_x = (float*)((char*)d_ws + byte_off);  byte_off += HD * sizeof(float);
    float* inv_e = (float*)((char*)d_ws + byte_off);  byte_off += HD * sizeof(float);
    float* bR[13];
    for (int i = 0; i < 13; ++i) {
        bR[i] = (float*)((char*)d_ws + byte_off);
        byte_off += HD * sizeof(float);
    }
    byte_off = (byte_off + 511) & ~size_t(511);
    _Float16* hbuf = (_Float16*)((char*)d_ws + byte_off);

    // Weight conversion/transpose + inv-std + bias reorg (tiny; L2-resident).
    auto prep = [&](const float* W, _Float16* Wt, int din, int dout) {
        int total = din * dout;
        gnn_prep_w<<<(total + 255) / 256, 256, 0, stream>>>(W, Wt, din, dout);
    };
    for (int i = 0; i < 3; ++i) prep(ne_W[i], t_ne[i], HD, HD);
    for (int i = 0; i < 3; ++i) prep(ee_W[i], t_ee[i], HD, HD);
    prep(em_W[0], t_em[0], 384, HD);
    for (int i = 1; i < 4; ++i) prep(em_W[i], t_em[i], HD, HD);
    for (int i = 0; i < 3; ++i) prep(dc_W[i], t_dc[i], HD, HD);
    gnn_prep_inv<<<1, HD, 0, stream>>>(std_x, inv_x, HD);
    gnn_prep_inv<<<1, HD, 0, stream>>>(std_e, inv_e, HD);
    const float* rawb[13] = {ne_b[0], ne_b[1], ne_b[2], ee_b[0], ee_b[1], ee_b[2],
                             em_b[0], em_b[1], em_b[2], em_b[3],
                             dc_b[0], dc_b[1], dc_b[2]};
    for (int i = 0; i < 13; ++i)
        gnn_prep_bias<<<1, HD, 0, stream>>>(rawb[i], bR[i]);

    // Node encoder
    NodeParams np;
    np.x = x; np.mean = mean_x; np.inv = inv_x;
    np.w0 = t_ne[0]; np.w1 = t_ne[1]; np.w2 = t_ne[2];
    np.b0 = bR[0]; np.b1 = bR[1]; np.b2 = bR[2];
    np.gamma = ne_g; np.beta = ne_bt; np.h = hbuf; np.N = N;
    {
        int slabs = (N + 15) / 16;
        int blocks = (slabs + NW - 1) / NW;
        gnn_node_kernel<<<blocks, BLK, 0, stream>>>(np);
    }

    // Fused edge pipeline
    EdgeParams ep;
    ep.eidx = eidx; ep.eattr = eattr; ep.mean_e = mean_e; ep.inv_e = inv_e;
    ep.h = hbuf;
    ep.ee0 = t_ee[0]; ep.ee1 = t_ee[1]; ep.ee2 = t_ee[2];
    ep.em0 = t_em[0]; ep.em1 = t_em[1]; ep.em2 = t_em[2]; ep.em3 = t_em[3];
    ep.dc0 = t_dc[0]; ep.dc1 = t_dc[1]; ep.dc2 = t_dc[2];
    ep.bee0 = bR[3]; ep.bee1 = bR[4]; ep.bee2 = bR[5];
    ep.bem0 = bR[6]; ep.bem1 = bR[7]; ep.bem2 = bR[8]; ep.bem3 = bR[9];
    ep.bdc0 = bR[10]; ep.bdc1 = bR[11]; ep.bdc2 = bR[12];
    ep.g_ee = ee_g; ep.bt_ee = ee_bt; ep.g_em = em_g; ep.bt_em = em_bt;
    ep.out = (float*)d_out; ep.E = E;
    {
        int slabs = (E + 15) / 16;
        int blocks = (slabs + NW - 1) / NW;
        gnn_edge_kernel<<<blocks, BLK, 0, stream>>>(ep);
    }
}
